// MultiHeadAttention_84138409329318
// MI455X (gfx1250) — matrix operand order repaired and checked by reading, not compiled, not run on hardware
//
#include <hip/hip_runtime.h>
#include <math.h>

// Problem constants (fixed by the reference).
#define DMDL   1024
#define NHEAD  16
#define DKH    64
#define NBAT   4
#define TSEQ   2048
#define MROWS  (NBAT * TSEQ)   // 8192
#define MB     (1024ull * 1024ull)

typedef __attribute__((ext_vector_type(4)))  _Float16 v4h;
typedef __attribute__((ext_vector_type(8)))  _Float16 v8h;
typedef __attribute__((ext_vector_type(16))) _Float16 v16h;
typedef __attribute__((ext_vector_type(8)))  float    v8f;
typedef __attribute__((ext_vector_type(4)))  unsigned int v4u;
typedef __attribute__((ext_vector_type(8)))  int      v8i;
typedef __attribute__((ext_vector_type(4)))  int      v4i;

#if __has_builtin(__builtin_amdgcn_tensor_load_to_lds)
#define HAVE_TDM 1
#else
#define HAVE_TDM 0
#endif

static __device__ __forceinline__ void wait_tensorcnt0() {
#if __has_builtin(__builtin_amdgcn_s_wait_tensorcnt)
  __builtin_amdgcn_s_wait_tensorcnt(0);
#else
  asm volatile("s_wait_tensorcnt 0x0" ::: "memory");
#endif
}

static __device__ __forceinline__ void wait_tensorcnt2() {
#if __has_builtin(__builtin_amdgcn_s_wait_tensorcnt)
  __builtin_amdgcn_s_wait_tensorcnt(2);
#else
  asm volatile("s_wait_tensorcnt 0x2" ::: "memory");
#endif
}

#if HAVE_TDM
// 2D tile DMA: global f16 tensor -> LDS, with optional per-line LDS padding.
//   tensor_d0/d1 : tensor extents (elements), d0 = contiguous dim
//   tile_d0/d1   : tile extents (elements)
//   d0_stride    : tensor line stride (elements)
//   pad codes    : pad_interval 0=2dw..7=256dw ; pad_amount 0=1dw..127=128dw
static __device__ __forceinline__ void tdm_load_2d(
    const _Float16* gaddr, unsigned int lds_byte_off,
    unsigned int tensor_d0, unsigned int tensor_d1,
    unsigned int tile_d0, unsigned int tile_d1,
    unsigned long long d0_stride,
    unsigned int pad_interval_code, unsigned int pad_amount_code, bool pad_en)
{
  unsigned long long ga = (unsigned long long)(uintptr_t)gaddr;
  v4u g0;
  g0[0] = 1u;                                   // count=1, user descriptor
  g0[1] = lds_byte_off;                         // lds_addr
  g0[2] = (unsigned int)ga;                     // global_addr[31:0]
  g0[3] = (unsigned int)(ga >> 32) | (2u << 30);// global_addr[56:32] | type=2
  unsigned int w0 = (1u << 16);                 // data_size = 2 bytes
  if (pad_en)
    w0 |= (1u << 20) | (pad_interval_code << 22) | (pad_amount_code << 25);
  v8i g1;
  g1[0] = (int)w0;
  g1[1] = (int)((tensor_d0 & 0xFFFFu) << 16);   // abar_addr=0 | dim0 lo16
  g1[2] = (int)((tensor_d0 >> 16) | ((tensor_d1 & 0xFFFFu) << 16));
  g1[3] = (int)((tensor_d1 >> 16) | (tile_d0 << 16));
  g1[4] = (int)(tile_d1 & 0xFFFFu);             // tile_dim2 = 0
  g1[5] = (int)(unsigned int)(d0_stride & 0xFFFFFFFFull);
  g1[6] = (int)(unsigned int)((d0_stride >> 32) & 0xFFFFull); // dim1_stride=0
  g1[7] = 0;
  v4i z4; z4[0] = 0; z4[1] = 0; z4[2] = 0; z4[3] = 0;
  v8i z8;
#pragma unroll
  for (int i = 0; i < 8; ++i) z8[i] = 0;
  __builtin_amdgcn_tensor_load_to_lds(g0, g1, z4, z4, z8, 0);
}
#endif

static __device__ __forceinline__ v16h make_v16h(v8h lo, v8h hi) {
  union { v16h v; v8h h[2]; } u;
  u.h[0] = lo; u.h[1] = hi;
  return u.v;
}

static __device__ __forceinline__ v8f zero_v8f() {
  v8f z;
#pragma unroll
  for (int i = 0; i < 8; ++i) z[i] = 0.0f;
  return z;
}

static __device__ __forceinline__ v8f wmma_f16f32(v16h a, v16h b, v8f c) {
  return __builtin_amdgcn_wmma_f32_16x16x32_f16(
      false, a, false, b, (short)0, c, false, false);
}

// ---------------------------------------------------------------------------
// fp32 -> f16 conversion (vectorized float4 -> 4x f16)
// ---------------------------------------------------------------------------
__global__ __launch_bounds__(256) void cvt_f32_f16(
    const float* __restrict__ s, _Float16* __restrict__ d, int n4)
{
  int i = blockIdx.x * 256 + threadIdx.x;
  if (i < n4) {
    const float4 x = ((const float4*)s)[i];
    v4h y;
    y[0] = (_Float16)x.x; y[1] = (_Float16)x.y;
    y[2] = (_Float16)x.z; y[3] = (_Float16)x.w;
    ((v4h*)d)[i] = y;
  }
}

// ---------------------------------------------------------------------------
// fp32 [rows][cols] -> f16 transposed [cols][rows], 32x32 LDS tiles.
// ---------------------------------------------------------------------------
__global__ __launch_bounds__(256) void cvt_transpose_f32_f16(
    const float* __restrict__ S, _Float16* __restrict__ D, int rows, int cols)
{
  __shared__ float tile[32][33];
  const int bx = blockIdx.x * 32;      // col base
  const int by = blockIdx.y * 32;      // row base
  const int tx = threadIdx.x & 31;
  const int ty = threadIdx.x >> 5;     // 0..7
#pragma unroll
  for (int r = 0; r < 32; r += 8)
    tile[r + ty][tx] = S[(size_t)(by + r + ty) * cols + bx + tx];
  __syncthreads();
#pragma unroll
  for (int r = 0; r < 32; r += 8)
    D[(size_t)(bx + r + ty) * rows + by + tx] = (_Float16)tile[tx][r + ty];
}

// ---------------------------------------------------------------------------
// GEMM: Out = X[M,K](f16) * W[K,N](f16, supplied pre-transposed WT=[N][K])
//       + bias[N](fp32)
//   OUT_MODE: 0 -> f16 head-split  [n][h][t][dk]   (Q,K)
//             1 -> f16 head-split transposed [n][h][dk][t]  (V)
//             2 -> fp32 row-major [M,N]             (final output)
// Block 256 = 8 waves; tile 128x128; K-tile 32.
// Both A and B tiles via double-buffered TDM (HW-padded to pitch 40);
// DMA of tile i+1 overlaps WMMA of tile i (s_wait_tensorcnt 2).
// ---------------------------------------------------------------------------
template <int OUT_MODE>
__global__ __launch_bounds__(256) void gemm_f16_wmma(
    const _Float16* __restrict__ Xh, const _Float16* __restrict__ WT,
    const float* __restrict__ bias, void* __restrict__ Outv,
    int M, int N, int K)
{
  constexpr int LDT = 40;                        // 32 + 8 pad halfwords
  constexpr unsigned TILE_B = 128 * LDT * 2;     // 10240 bytes per tile
  extern __shared__ _Float16 smem[];             // dynamic => LDS offset 0
  // layout: bufA0 | bufB0 | bufA1 | bufB1

  const int tid  = threadIdx.x;
  const int lane = tid & 31;
  const int wid  = tid >> 5;
  const int wm   = wid >> 1;
  const int wn   = wid & 1;
  const int lr   = lane & 15;
  const int lh   = lane >> 4;

  const int m0 = blockIdx.x * 128;
  const int n0 = blockIdx.y * 128;

  v8f acc[2][4];
#pragma unroll
  for (int i = 0; i < 2; ++i)
#pragma unroll
    for (int j = 0; j < 4; ++j) acc[i][j] = zero_v8f();

  const int NT = K / 32;

#if HAVE_TDM
  if (wid == 0) {   // prologue: tile 0 into buffer 0
    tdm_load_2d(Xh + (size_t)m0 * K, 0,
                (unsigned)K, (unsigned)M, 32u, 128u, (unsigned long long)K,
                3u, 3u, true);
    tdm_load_2d(WT + (size_t)n0 * K, TILE_B,
                (unsigned)K, (unsigned)N, 32u, 128u, (unsigned long long)K,
                3u, 3u, true);
  }
#endif

  for (int it = 0; it < NT; ++it) {
    const int b = it & 1;
    _Float16* curA = smem + (b ? (2 * TILE_B / 2) : 0);
    _Float16* curB = curA + TILE_B / 2;

    __syncthreads();   // all reads of the other buffer (iter it-1) retired
#if HAVE_TDM
    if (wid == 0) {
      if (it + 1 < NT) {
        const int ktn = (it + 1) * 32;
        const unsigned nb = (b ^ 1) ? (2 * TILE_B) : 0;
        tdm_load_2d(Xh + (size_t)m0 * K + ktn, nb,
                    (unsigned)K, (unsigned)M, 32u, 128u, (unsigned long long)K,
                    3u, 3u, true);
        tdm_load_2d(WT + (size_t)n0 * K + ktn, nb + TILE_B,
                    (unsigned)K, (unsigned)N, 32u, 128u, (unsigned long long)K,
                    3u, 3u, true);
        wait_tensorcnt2();   // tile it resident; tile it+1 in flight
      } else {
        wait_tensorcnt0();
      }
    }
#else
    {
      const int kt = it * 32;
#pragma unroll
      for (int itr = 0; itr < 2; ++itr) {
        int idx = tid + itr * 256;                 // 0..511
        int r = idx >> 2, c8 = (idx & 3) * 8;
        *(v8h*)&curA[r * LDT + c8] =
            *(const v8h*)&Xh[(size_t)(m0 + r) * K + kt + c8];
        *(v8h*)&curB[r * LDT + c8] =
            *(const v8h*)&WT[(size_t)(n0 + r) * K + kt + c8];
      }
    }
#endif
    __syncthreads();

    v16h afr[2];
#pragma unroll
    for (int i = 0; i < 2; ++i) {
      const _Float16* ap = &curA[(wm * 32 + i * 16 + lr) * LDT];
      afr[i] = make_v16h(*(const v8h*)(ap + lh * 8),
                         *(const v8h*)(ap + 16 + lh * 8));
    }
    v16h bfr[4];
#pragma unroll
    for (int j = 0; j < 4; ++j) {
      const _Float16* bp = &curB[(wn * 64 + j * 16 + lr) * LDT + lh * 8];
      bfr[j] = make_v16h(*(const v8h*)bp, *(const v8h*)(bp + 16));
    }
#pragma unroll
    for (int i = 0; i < 2; ++i)
#pragma unroll
      for (int j = 0; j < 4; ++j)
        acc[i][j] = wmma_f16f32(afr[i], bfr[j], acc[i][j]);
  }

  // ---- epilogue: C layout = lane col (lr), rows v + 8*lh ----
#pragma unroll
  for (int i = 0; i < 2; ++i) {
#pragma unroll
    for (int j = 0; j < 4; ++j) {
      const int Rl = wm * 32 + i * 16 + lh * 8;
      const int C  = n0 + wn * 64 + j * 16 + lr;
      const float bC = bias[C];
#pragma unroll
      for (int v = 0; v < 8; ++v) {
        const int R = m0 + Rl + v;
        const float val = acc[i][j][v] + bC;
        if constexpr (OUT_MODE == 0) {
          const int nb = R >> 11, t = R & (TSEQ - 1);
          const int hh = C >> 6,  d = C & (DKH - 1);
          ((_Float16*)Outv)[(((size_t)(nb * NHEAD + hh)) * TSEQ + t) * DKH + d] =
              (_Float16)val;
        } else if constexpr (OUT_MODE == 1) {
          const int nb = R >> 11, t = R & (TSEQ - 1);
          const int hh = C >> 6,  d = C & (DKH - 1);
          ((_Float16*)Outv)[(((size_t)(nb * NHEAD + hh)) * DKH + d) * TSEQ + t] =
              (_Float16)val;
        } else {
          ((float*)Outv)[(size_t)R * N + C] = val;
        }
      }
    }
  }
}

// ---------------------------------------------------------------------------
// Flash attention (causal + key pad mask).
//   Qh,Kh: [n][h][t][64] f16, Vt: [n][h][64][t] f16, Ao: [n][t][h*64+d] f16
// Block = 8 waves sharing one (n,h); K/V key-block tiles via double-buffered
// TDM into block LDS (HW-padded); each wave owns a 16-row Q tile.
// ---------------------------------------------------------------------------
__global__ __launch_bounds__(256) void flash_attn_wmma(
    const _Float16* __restrict__ Qh, const _Float16* __restrict__ Kh,
    const _Float16* __restrict__ Vt, const int* __restrict__ pad,
    _Float16* __restrict__ Ao)
{
  constexpr int LDK = 72;   // 64 + 8 pad (TDM: 4dw per 32dw line)
  constexpr int LDV = 40;   // 32 + 8 pad (TDM: 4dw per 16dw line)
  constexpr int LP  = 40;
  constexpr unsigned KT_B = 32 * LDK * 2;            // 4608 B per K tile
  constexpr unsigned VT_B = 64 * LDV * 2;            // 5120 B per V tile
  constexpr unsigned K0_OFF = 0;
  constexpr unsigned K1_OFF = KT_B;                  // 4608
  constexpr unsigned V0_OFF = 2 * KT_B;              // 9216
  constexpr unsigned V1_OFF = 2 * KT_B + VT_B;       // 14336
  constexpr unsigned P_OFF  = 2 * KT_B + 2 * VT_B;   // 19456
  extern __shared__ _Float16 smem[];

  const int tid  = threadIdx.x;
  const int lane = tid & 31;
  const int wid  = tid >> 5;
  const int lr   = lane & 15;
  const int lh   = lane >> 4;

  const int nh = blockIdx.y;
  const int n  = nh >> 4;
  const int h  = nh & 15;
  const int q0 = blockIdx.x * 128 + wid * 16;

  const _Float16* Qp = Qh + (size_t)nh * TSEQ * DKH;
  const _Float16* Kp = Kh + (size_t)nh * TSEQ * DKH;
  const _Float16* Vp = Vt + (size_t)nh * DKH * TSEQ;
  _Float16* lp = smem + P_OFF / 2 + wid * 16 * LP;

  // Q A-fragments (16x64), straight from global.
  v16h aQ[2];
  {
    const _Float16* qrow = Qp + (size_t)(q0 + lr) * DKH;
#pragma unroll
    for (int ks = 0; ks < 2; ++ks)
      aQ[ks] = make_v16h(*(const v8h*)(qrow + ks * 32 + lh * 8),
                         *(const v8h*)(qrow + ks * 32 + 16 + lh * 8));
  }

  float mrow[8], lsum[8];
  v8f accO[4];
#pragma unroll
  for (int v = 0; v < 8; ++v) { mrow[v] = -1e30f; lsum[v] = 0.0f; }
#pragma unroll
  for (int c = 0; c < 4; ++c) accO[c] = zero_v8f();

  const int kb_blk_max = (blockIdx.x * 128 + 127) / 32;  // causal bound (block)

#if HAVE_TDM
  if (wid == 0) {   // prologue: key-block 0 into buffer 0
    tdm_load_2d(Kp, K0_OFF, 64u, (unsigned)TSEQ, 64u, 32u, 64ull, 4u, 3u, true);
    tdm_load_2d(Vp, V0_OFF, (unsigned)TSEQ, 64u, 32u, 64u,
                (unsigned long long)TSEQ, 3u, 3u, true);
  }
#endif

  for (int kb = 0; kb <= kb_blk_max; ++kb) {
    const int kbase = kb * 32;
    const int b = kb & 1;
    _Float16* curK = smem + (b ? K1_OFF : K0_OFF) / 2;
    _Float16* curV = smem + (b ? V1_OFF : V0_OFF) / 2;

    __syncthreads();   // reads of other buffer retired
#if HAVE_TDM
    if (wid == 0) {
      if (kb + 1 <= kb_blk_max) {
        const int kn = (kb + 1) * 32;
        tdm_load_2d(Kp + (size_t)kn * DKH, b ? K0_OFF : K1_OFF,
                    64u, (unsigned)TSEQ, 64u, 32u, 64ull, 4u, 3u, true);
        tdm_load_2d(Vp + kn, b ? V0_OFF : V1_OFF,
                    (unsigned)TSEQ, 64u, 32u, 64u,
                    (unsigned long long)TSEQ, 3u, 3u, true);
        wait_tensorcnt2();
      } else {
        wait_tensorcnt0();
      }
    }
#else
    {
      int idx = tid;                              // K: 32x64/8 = 256 v8h
      int r = idx >> 3, c8 = (idx & 7) * 8;
      *(v8h*)&curK[r * LDK + c8] =
          *(const v8h*)&Kp[(size_t)(kbase + r) * DKH + c8];
      r = idx >> 2; c8 = (idx & 3) * 8;           // V: 64x32/8 = 256 v8h
      *(v8h*)&curV[r * LDV + c8] =
          *(const v8h*)&Vp[(size_t)r * TSEQ + kbase + c8];
    }
#endif
    __syncthreads();

    if (kbase <= q0 + 15) {
      // K^T B-fragments from LDS.
      v16h bK[2][2];
#pragma unroll
      for (int sub = 0; sub < 2; ++sub) {
        const _Float16* krow = &curK[(sub * 16 + lr) * LDK];
#pragma unroll
        for (int ks = 0; ks < 2; ++ks) {
          const _Float16* kp2 = krow + ks * 32 + lh * 8;
          bK[sub][ks] = make_v16h(*(const v8h*)kp2, *(const v8h*)(kp2 + 16));
        }
      }

      float Sf[2][8];
      float tmax[8];
#pragma unroll
      for (int v = 0; v < 8; ++v) tmax[v] = -1e30f;

#pragma unroll
      for (int sub = 0; sub < 2; ++sub) {
        v8f s = wmma_f16f32(aQ[0], bK[sub][0], zero_v8f());
        s = wmma_f16f32(aQ[1], bK[sub][1], s);
        const int key = kbase + sub * 16 + lr;
        const int pm  = pad[n * TSEQ + key];
#pragma unroll
        for (int v = 0; v < 8; ++v) {
          const int row = q0 + v + lh * 8;
          float sv = s[v] * 0.125f;               // 1/sqrt(64)
          if (key > row || pm == 0) sv = -1e30f;
          Sf[sub][v] = sv;
          tmax[v] = fmaxf(tmax[v], sv);
        }
      }

#pragma unroll
      for (int off = 1; off < 16; off <<= 1)
#pragma unroll
        for (int v = 0; v < 8; ++v)
          tmax[v] = fmaxf(tmax[v], __shfl_xor(tmax[v], off, 16));

      float alpha[8], rsum[8];
#pragma unroll
      for (int v = 0; v < 8; ++v) {
        const float nm = fmaxf(mrow[v], tmax[v]);
        alpha[v] = __expf(mrow[v] - nm);
        mrow[v]  = nm;
        rsum[v]  = 0.0f;
      }

#pragma unroll
      for (int sub = 0; sub < 2; ++sub)
#pragma unroll
        for (int v = 0; v < 8; ++v) {
          float pv = __expf(Sf[sub][v] - mrow[v]);
          if (Sf[sub][v] <= -1e29f) pv = 0.0f;
          rsum[v] += pv;
          lp[(v + lh * 8) * LP + sub * 16 + lr] = (_Float16)pv;
        }

#pragma unroll
      for (int off = 1; off < 16; off <<= 1)
#pragma unroll
        for (int v = 0; v < 8; ++v)
          rsum[v] += __shfl_xor(rsum[v], off, 16);

#pragma unroll
      for (int v = 0; v < 8; ++v) lsum[v] = lsum[v] * alpha[v] + rsum[v];
#pragma unroll
      for (int c = 0; c < 4; ++c)
#pragma unroll
        for (int v = 0; v < 8; ++v) accO[c][v] *= alpha[v];

      // P back as A-fragment (same-wave DS ops are in-order).
      v16h aP;
      {
        const _Float16* pr = lp + lr * LP;
        aP = make_v16h(*(const v8h*)(pr + lh * 8),
                       *(const v8h*)(pr + 16 + lh * 8));
      }

      // O += P * V from LDS V tile.
#pragma unroll
      for (int c = 0; c < 4; ++c) {
        const _Float16* vp2 = &curV[(c * 16 + lr) * LDV + lh * 8];
        v16h bV = make_v16h(*(const v8h*)vp2, *(const v8h*)(vp2 + 16));
        accO[c] = wmma_f16f32(aP, bV, accO[c]);
      }
    }
  }

  // Write normalized O as row-major [N*T, H*DK] f16.
#pragma unroll
  for (int c = 0; c < 4; ++c)
#pragma unroll
    for (int v = 0; v < 8; ++v) {
      const float val = accO[c][v] / lsum[v];
      const int row = q0 + v + lh * 8;
      const int d   = c * 16 + lr;
      Ao[((size_t)(n * TSEQ + row)) * DMDL + h * DKH + d] = (_Float16)val;
    }
}

// ---------------------------------------------------------------------------
extern "C" void kernel_launch(void* const* d_in, const int* in_sizes, int n_in,
                              void* d_out, int out_size, void* d_ws, size_t ws_size,
                              hipStream_t stream)
{
  const float* q   = (const float*)d_in[0];
  const float* k   = (const float*)d_in[1];
  const float* v   = (const float*)d_in[2];
  const int*   pad = (const int*)  d_in[3];
  const float* Wq  = (const float*)d_in[4];
  const float* bq  = (const float*)d_in[5];
  const float* Wk  = (const float*)d_in[6];
  const float* bk  = (const float*)d_in[7];
  const float* Wv  = (const float*)d_in[8];
  const float* bv  = (const float*)d_in[9];
  const float* Wo  = (const float*)d_in[10];
  const float* bo  = (const float*)d_in[11];

  char* ws = (char*)d_ws;
  _Float16* Xq  = (_Float16*)(ws);                 // 16 MB each
  _Float16* Xk  = (_Float16*)(ws + 16 * MB);
  _Float16* Xv  = (_Float16*)(ws + 32 * MB);
  _Float16* WTq = (_Float16*)(ws + 48 * MB);       // 2 MB each, [N][K] f16
  _Float16* WTk = (_Float16*)(ws + 50 * MB);
  _Float16* WTv = (_Float16*)(ws + 52 * MB);
  _Float16* WTo = (_Float16*)(ws + 54 * MB);
  _Float16* Qh  = (_Float16*)(ws + 56 * MB);       // 16 MB each
  _Float16* Kh  = (_Float16*)(ws + 72 * MB);
  _Float16* Vt  = (_Float16*)(ws + 88 * MB);
  _Float16* Ao  = (_Float16*)(ws + 104 * MB);

  // fp32 -> f16 conversion (activations) and transpose-convert (weights).
  const int nX4 = MROWS * DMDL / 4;
  cvt_f32_f16<<<(nX4 + 255) / 256, 256, 0, stream>>>(q, Xq, nX4);
  cvt_f32_f16<<<(nX4 + 255) / 256, 256, 0, stream>>>(k, Xk, nX4);
  cvt_f32_f16<<<(nX4 + 255) / 256, 256, 0, stream>>>(v, Xv, nX4);
  const dim3 gt(DMDL / 32, DMDL / 32);
  cvt_transpose_f32_f16<<<gt, 256, 0, stream>>>(Wq, WTq, DMDL, DMDL);
  cvt_transpose_f32_f16<<<gt, 256, 0, stream>>>(Wk, WTk, DMDL, DMDL);
  cvt_transpose_f32_f16<<<gt, 256, 0, stream>>>(Wv, WTv, DMDL, DMDL);
  cvt_transpose_f32_f16<<<gt, 256, 0, stream>>>(Wo, WTo, DMDL, DMDL);

  const dim3 gg(MROWS / 128, DMDL / 128);          // 64 x 8
  const size_t gemm_lds = (size_t)4 * 128 * 40 * sizeof(_Float16);  // 40960 B
  gemm_f16_wmma<0><<<gg, 256, gemm_lds, stream>>>(Xq, WTq, bq, Qh, MROWS, DMDL, DMDL);
  gemm_f16_wmma<0><<<gg, 256, gemm_lds, stream>>>(Xk, WTk, bk, Kh, MROWS, DMDL, DMDL);
  gemm_f16_wmma<1><<<gg, 256, gemm_lds, stream>>>(Xv, WTv, bv, Vt, MROWS, DMDL, DMDL);

  const size_t attn_lds =
      (size_t)(2 * 32 * 72 + 2 * 64 * 40 + 8 * 16 * 40) * sizeof(_Float16);  // 29696 B
  flash_attn_wmma<<<dim3(TSEQ / 128, NBAT * NHEAD), 256, attn_lds, stream>>>(
      Qh, Kh, Vt, pad, Ao);

  gemm_f16_wmma<2><<<gg, 256, gemm_lds, stream>>>(Ao, WTo, bo, d_out, MROWS, DMDL, DMDL);
}
